// StandardMACE_21157008900120
// MI455X (gfx1250) — compile-verified, matrix operand order checked by reading, not run on hardware
//
#include <hip/hip_runtime.h>
#include <hip/hip_bf16.h>

// ---------------------------------------------------------------------------
// MACE forward + forces on MI455X (gfx1250).
//  * Radial MLP (8->64->64->64->128, E=160k edges) on v_wmma_f32_16x16x32_f16,
//    16-edge tiles per wave32. Weights pre-permuted into B-tile lane order in
//    global ws, then staged to LDS with global_load_async_to_lds_b128.
//  * A-tiles loaded as two 16B vector LDS reads (layout is two contiguous
//    8-half runs per lane); SiLU uses native v_rcp_f32.
//  * msg (N*K*16 = 16MB) accumulated with f32 global atomics (L2-resident).
//  * Energy depends only on the l=0 (Y0==1) channel and is linear after msg,
//    so forces = per-node seed gmsg0 + recompute-based per-edge backward.
// ---------------------------------------------------------------------------

typedef __attribute__((ext_vector_type(16))) _Float16 v16h;
typedef __attribute__((ext_vector_type(8)))  _Float16 v8h;
typedef __attribute__((ext_vector_type(8)))  float    v8f;

#define K_CH 32
#define NB   8
#define R_MAXF 5.0f
#define AVG_NEIGHF 20.0f
#define SCALE_C 0.8f
#define SHIFT_C 0.1f
#define TILE 16
#define WPB  4      // waves per block in edge_forward
#define BW_WAVES 8  // waves (edges) per block in edge_backward
#define PI_F 3.14159265358979f

// permuted f16 weight buffer layout (halves)
#define WH_L0_OFF 0
#define WH_L0_SZ  2048   // 4 tiles x 1 kstep (k padded 8->32)
#define WH_L1_OFF 2048
#define WH_L1_SZ  4096   // 4 tiles x 2 ksteps
#define WH_L2_OFF 6144
#define WH_L2_SZ  4096
#define WH_L3_OFF 10240
#define WH_L3_SZ  8192   // 8 tiles x 2 ksteps
#define WH_TOTAL  18432

__constant__ int c_lof[16] = {0,1,1,1,2,2,2,2,2,3,3,3,3,3,3,3};

#if __has_builtin(__builtin_amdgcn_rcpf)
#define FAST_RCP(x) __builtin_amdgcn_rcpf(x)
#else
#define FAST_RCP(x) (1.f/(x))
#endif

__device__ inline float siluf(float x){
  float s = FAST_RCP(1.f + __expf(-x));
  return x*s;
}
__device__ inline float dsiluf(float x){
  float s = FAST_RCP(1.f + __expf(-x));
  return s*(1.f + x*(1.f - s));
}

// CDNA5 async global->LDS copy (16B per lane), tracked by ASYNCcnt.
// ldsoff = byte offset of destination inside the dynamic-LDS block (these
// kernels have no static LDS, so dynamic LDS starts at offset 0).
__device__ inline void async_copy_b128(const void* gsrc, unsigned ldsoff){
  unsigned long long ga = (unsigned long long)gsrc;
  asm volatile("global_load_async_to_lds_b128 %0, %1, off"
               :: "v"(ldsoff), "v"(ga) : "memory");
}
__device__ inline void async_wait(){
  asm volatile("s_wait_asynccnt 0x0" ::: "memory");
}

// k-index mapping for the CDNA5 16-bit 16x32 A/B tile VGPR layout:
// VGPR v holds a K pair; lanes 0-15 K in {0..7,16..23}, lanes 16-31 {8..15,24..31}
__device__ __host__ inline int wmma_k_of(int v, int hi){
  int base = (v < 4) ? (2*v) : (16 + 2*(v-4));
  return base + hi*8;
}

// A-tile: per lane the 16 halves are two contiguous 8-half runs of the row.
__device__ inline v16h load_tileA(const _Float16* act, int m, int hi, int kbase){
  const _Float16* row = act + m*64 + kbase + hi*8;
  v8h lo = *(const v8h*)(row);        // k = kbase+hi*8 .. +7   (VGPR 0..3)
  v8h hh = *(const v8h*)(row + 16);   // k = kbase+16+hi*8 .. +7 (VGPR 4..7)
  return __builtin_shufflevector(lo, hh, 0,1,2,3,4,5,6,7,8,9,10,11,12,13,14,15);
}

// B-tile from the pre-permuted buffer: one contiguous 32B chunk per lane.
__device__ inline v16h load_tileB(const _Float16* wl, int tile, int lane){
  return *(const v16h*)(wl + ((size_t)tile*32 + lane)*16);
}

// ------------------------------- kernels -----------------------------------

__global__ void zero_f32(float* p, long n){
  long i = (long)blockIdx.x*blockDim.x + threadIdx.x;
  long stride = (long)gridDim.x*blockDim.x;
  for (; i < n; i += stride) p[i] = 0.f;
}

// Pre-permute MLP weights into WMMA B-tile lane order (f16) + extract the
// m=0 columns of rW3 (f32) used by the backward pass.
__global__ void prep_weights(const float* __restrict__ rW0, const float* __restrict__ rW1,
                             const float* __restrict__ rW2, const float* __restrict__ rW3,
                             _Float16* __restrict__ wH, float* __restrict__ w3c){
  int i = blockIdx.x*blockDim.x + threadIdx.x;
  int stride = gridDim.x*blockDim.x;
  for (int t = i; t < WH_TOTAL; t += stride){
    const float* W; int ld, krows, ks_cnt, rel;
    if (t < WH_L1_OFF)      { W=rW0; ld=64;  krows=8;  ks_cnt=1; rel=t; }
    else if (t < WH_L2_OFF) { W=rW1; ld=64;  krows=64; ks_cnt=2; rel=t-WH_L1_OFF; }
    else if (t < WH_L3_OFF) { W=rW2; ld=64;  krows=64; ks_cnt=2; rel=t-WH_L2_OFF; }
    else                    { W=rW3; ld=128; krows=64; ks_cnt=2; rel=t-WH_L3_OFF; }
    int tile = rel >> 9;          // /512 halves per tile
    int lane = (rel >> 4) & 31;
    int v2   = rel & 15;
    int v = v2 >> 1, odd = v2 & 1;
    int nt = tile / ks_cnt, ks = tile % ks_cnt;
    int hi = lane >> 4;
    int n = nt*16 + (lane & 15);
    int k = wmma_k_of(v, hi) + ks*32 + odd;
    wH[t] = (k < krows) ? (_Float16)W[k*ld + n] : (_Float16)0.f;
  }
  for (int t = i; t < 2048; t += stride){
    int j = t >> 5, k = t & 31;
    w3c[t] = rW3[j*128 + 4*k];
  }
}

// nodeUp = (attrs @ W_embed) @ W_up ; zidx = one-hot index ; e0 per node
__global__ void node_prep(const float* __restrict__ attrs, const float* __restrict__ W_embed,
                          const float* __restrict__ W_up, const float* __restrict__ atomic_e,
                          float* __restrict__ nodeUp, int* __restrict__ zidx,
                          float* __restrict__ nodeE0, int N, int Z){
  __shared__ float embL[K_CH];
  int n = blockIdx.x; int lane = threadIdx.x;
  float acc = 0.f;
  for (int z = 0; z < Z; ++z) acc += attrs[(long)n*Z+z]*W_embed[z*K_CH+lane];
  embL[lane] = acc;
  __syncthreads();
  float up = 0.f;
  for (int j = 0; j < K_CH; ++j) up += embL[j]*W_up[j*K_CH+lane];
  nodeUp[(long)n*K_CH+lane] = up;
  if (lane == 0){
    int zi = 0;
    for (int z = 0; z < Z; ++z) if (attrs[(long)n*Z+z] > 0.5f) zi = z;
    zidx[n] = zi;
    nodeE0[n] = atomic_e[zi];
  }
}

// Per-edge forward: geometry + bessel + Y + radial MLP (WMMA) + mji scatter.
__global__ void edge_forward(const float* __restrict__ pos, const float* __restrict__ shifts,
                             const int* __restrict__ ei,
                             const _Float16* __restrict__ wH,
                             const float* __restrict__ nodeUp, float* __restrict__ msg, int E){
  extern __shared__ char smem[];
  const int wave = threadIdx.x >> 5;
  const int lane = threadIdx.x & 31;
  const int m = lane & 15, hi = lane >> 4;

  // carve: [f32 per-wave: Y(256) hs(512) tp(2048)] [int recv per-wave 16]
  //        [f16: permuted weights (18432) ; acts per-wave 2*1024]
  float* f32base = (float*)smem;
  float* YL  = f32base + wave*2816;
  float* hsL = YL + 256;
  float* tpL = hsL + 512;
  int*   recvL = (int*)(smem + (size_t)WPB*2816*4) + wave*16;
  const unsigned hbaseOff = (unsigned)(WPB*2816*4 + WPB*16*4);
  _Float16* hbase = (_Float16*)(smem + hbaseOff);
  _Float16* wl0 = hbase + WH_L0_OFF;
  _Float16* wl1 = hbase + WH_L1_OFF;
  _Float16* wl2 = hbase + WH_L2_OFF;
  _Float16* wl3 = hbase + WH_L3_OFF;
  _Float16* actA = hbase + WH_TOTAL + wave*(2*16*64);
  _Float16* actB = actA + 16*64;

  // ---- async-stage the permuted weights into LDS (16B per lane-op) ----
  {
    const int chunks = WH_TOTAL/8;              // 2304 x 16B
    for (int i = threadIdx.x; i < chunks; i += blockDim.x)
      async_copy_b128(wH + (size_t)i*8, hbaseOff + (unsigned)i*16);
    async_wait();
  }

  const int tile  = blockIdx.x*WPB + wave;
  const int ebase = tile*TILE;
  int e  = ebase + m;
  int ec = (e < E) ? e : (E-1);
  int snd = ei[ec], rcv = ei[E + ec];
  if (lane == 0 && ebase + TILE < E) __builtin_prefetch(&ei[ebase + TILE], 0, 0);

  float vx = pos[(long)rcv*3+0]-pos[(long)snd*3+0]+shifts[(long)ec*3+0];
  float vy = pos[(long)rcv*3+1]-pos[(long)snd*3+1]+shifts[(long)ec*3+1];
  float vz = pos[(long)rcv*3+2]-pos[(long)snd*3+2]+shifts[(long)ec*3+2];
  float r  = fmaxf(sqrtf(vx*vx+vy*vy+vz*vz), 1e-6f);
  float inv = 1.f/r;
  float ux = vx*inv, uy = vy*inv, uz = vz*inv;

  if (hi == 0){
    recvL[m] = rcv;
    const float s3=1.73205081f, s5=2.23606798f, s15=3.87298335f;
    const float c1=2.09165007f, c2=10.2469508f, c3=1.62018517f, c4=1.32287566f, c5=5.12347538f;
    float z2 = uz*uz;
    YL[m*16+ 0]=1.f;
    YL[m*16+ 1]=s3*ux;  YL[m*16+ 2]=s3*uy;  YL[m*16+ 3]=s3*uz;
    YL[m*16+ 4]=s15*ux*uy; YL[m*16+ 5]=s15*uy*uz;
    YL[m*16+ 6]=0.5f*s5*(3.f*z2-1.f); YL[m*16+ 7]=s15*ux*uz;
    YL[m*16+ 8]=0.5f*s15*(ux*ux-uy*uy);
    YL[m*16+ 9]=c1*uy*(3.f*ux*ux-uy*uy); YL[m*16+10]=c2*ux*uy*uz;
    YL[m*16+11]=c3*uy*(5.f*z2-1.f);      YL[m*16+12]=c4*uz*(5.f*z2-3.f);
    YL[m*16+13]=c3*ux*(5.f*z2-1.f);      YL[m*16+14]=c5*uz*(ux*ux-uy*uy);
    YL[m*16+15]=c1*ux*(ux*ux-3.f*uy*uy);
    float x = r*(1.f/R_MAXF);
    float cut = 0.f;
    if (x < 1.f){
      float x5 = x*x*x*x*x;
      cut = 1.f - 21.f*x5 + 35.f*x5*x - 15.f*x5*x*x;
    }
    const float C = 0.632455532f; // sqrt(2/R_MAX)
#pragma unroll
    for (int j = 0; j < NB; ++j){
      float aj = (float)(j+1)*PI_F*(1.f/R_MAXF);
      actA[m*64+j] = (_Float16)(C*__sinf(aj*r)*inv*cut);
    }
    for (int j = NB; j < 64; ++j) actA[m*64+j] = (_Float16)0.f;
  }
  for (int kk = 0; kk < 16; ++kk) hsL[m*32 + hi*16 + kk] = nodeUp[(long)snd*32 + hi*16 + kk];
  __syncthreads();

  // ---- layer 0 : (16x8 padded to 32) x (8x64) ----
  {
    v16h a = load_tileA(actA, m, hi, 0);
#pragma unroll
    for (int nt = 0; nt < 4; ++nt){
      v16h b = load_tileB(wl0, nt, lane);
      v8f c = {};
      c = __builtin_amdgcn_wmma_f32_16x16x32_f16(false, a, false, b, (short)0, c, false, false);
#pragma unroll
      for (int rr = 0; rr < 8; ++rr)
        actB[(rr + hi*8)*64 + nt*16 + m] = (_Float16)siluf(c[rr]);
    }
  }
  __syncthreads();
  // ---- layer 1 : 64x64 ----
  {
    v16h a0 = load_tileA(actB, m, hi, 0);
    v16h a1 = load_tileA(actB, m, hi, 32);
#pragma unroll
    for (int nt = 0; nt < 4; ++nt){
      v8f c = {};
      v16h b0 = load_tileB(wl1, nt*2+0, lane);
      c = __builtin_amdgcn_wmma_f32_16x16x32_f16(false, a0, false, b0, (short)0, c, false, false);
      v16h b1 = load_tileB(wl1, nt*2+1, lane);
      c = __builtin_amdgcn_wmma_f32_16x16x32_f16(false, a1, false, b1, (short)0, c, false, false);
#pragma unroll
      for (int rr = 0; rr < 8; ++rr)
        actA[(rr + hi*8)*64 + nt*16 + m] = (_Float16)siluf(c[rr]);
    }
  }
  __syncthreads();
  // ---- layer 2 : 64x64 ----
  {
    v16h a0 = load_tileA(actA, m, hi, 0);
    v16h a1 = load_tileA(actA, m, hi, 32);
#pragma unroll
    for (int nt = 0; nt < 4; ++nt){
      v8f c = {};
      v16h b0 = load_tileB(wl2, nt*2+0, lane);
      c = __builtin_amdgcn_wmma_f32_16x16x32_f16(false, a0, false, b0, (short)0, c, false, false);
      v16h b1 = load_tileB(wl2, nt*2+1, lane);
      c = __builtin_amdgcn_wmma_f32_16x16x32_f16(false, a1, false, b1, (short)0, c, false, false);
#pragma unroll
      for (int rr = 0; rr < 8; ++rr)
        actB[(rr + hi*8)*64 + nt*16 + m] = (_Float16)siluf(c[rr]);
    }
  }
  __syncthreads();
  // ---- layer 3 : 64x128 (no activation) -> tp weights ----
  {
    v16h a0 = load_tileA(actB, m, hi, 0);
    v16h a1 = load_tileA(actB, m, hi, 32);
#pragma unroll
    for (int nt = 0; nt < 8; ++nt){
      v8f c = {};
      v16h b0 = load_tileB(wl3, nt*2+0, lane);
      c = __builtin_amdgcn_wmma_f32_16x16x32_f16(false, a0, false, b0, (short)0, c, false, false);
      v16h b1 = load_tileB(wl3, nt*2+1, lane);
      c = __builtin_amdgcn_wmma_f32_16x16x32_f16(false, a1, false, b1, (short)0, c, false, false);
#pragma unroll
      for (int rr = 0; rr < 8; ++rr)
        tpL[(rr + hi*8)*128 + nt*16 + m] = c[rr];
    }
  }
  __syncthreads();

  // ---- mji scatter: msg[rcv,k,m] += hsrc[k]*tp[4k+l(m)]*Y[m] ----
  const int k = lane;
  const int lof[16] = {0,1,1,1,2,2,2,2,2,3,3,3,3,3,3,3};
  for (int ee = 0; ee < TILE; ++ee){
    if (ebase + ee >= E) break;
    int rv = recvL[ee];
    float hv = hsL[ee*32 + k];
    float4 tp4 = *(const float4*)&tpL[ee*128 + 4*k];
    const float tpv[4] = {tp4.x, tp4.y, tp4.z, tp4.w};
    float4 y0 = *(const float4*)&YL[ee*16 + 0];
    float4 y1 = *(const float4*)&YL[ee*16 + 4];
    float4 y2 = *(const float4*)&YL[ee*16 + 8];
    float4 y3 = *(const float4*)&YL[ee*16 + 12];
    const float yv[16] = {y0.x,y0.y,y0.z,y0.w, y1.x,y1.y,y1.z,y1.w,
                          y2.x,y2.y,y2.z,y2.w, y3.x,y3.y,y3.z,y3.w};
    float* dst = msg + (size_t)rv*512 + k*16;
#pragma unroll
    for (int mm = 0; mm < 16; ++mm){
      atomicAdd(dst + mm, hv * tpv[lof[mm]] * yv[mm]);
    }
  }
}

// Per-node: per-l linears, skip contraction, products, readout, energies,
// plus gradient seed gmsg0[n,k] = dE/dmsg_raw[n,k,0].
__global__ void node_update(const float* __restrict__ msg, const int* __restrict__ zidx,
                            const float* __restrict__ nodeE0,
                            const float* __restrict__ W_lin, const float* __restrict__ W_skip,
                            const float* __restrict__ Wp_c, const float* __restrict__ Wp_lin,
                            const float* __restrict__ W_read, const int* __restrict__ batch,
                            float* __restrict__ outTotal, float* __restrict__ outNodeE,
                            float* __restrict__ outInter, float* __restrict__ outNF,
                            float* __restrict__ gmsg0, int N, int Z){
  __shared__ float msgL[32*16], msg2L[32*16], AL[32*16];
  __shared__ float ccL[2*32], t1L[32], t2L[32], redL[32];
  int n = blockIdx.x; int c = threadIdx.x;
  int z = zidx[n];
  for (int mm = 0; mm < 16; ++mm)
    msgL[c*16+mm] = msg[(size_t)n*512 + c*16 + mm] * (1.f/AVG_NEIGHF);
  ccL[c]    = Wp_c[(0*Z+z)*32 + c];
  ccL[32+c] = Wp_c[(1*Z+z)*32 + c];
  __syncthreads();
  for (int mm = 0; mm < 16; ++mm){
    int l = c_lof[mm];
    float acc = 0.f;
    for (int k = 0; k < 32; ++k) acc += msgL[k*16+mm]*W_lin[(l*32+k)*32 + c];
    msg2L[c*16+mm] = acc;
  }
  __syncthreads();
  for (int mm = 0; mm < 16; ++mm){
    int l = c_lof[mm];
    float acc = 0.f;
    for (int k = 0; k < 32; ++k) acc += msg2L[k*16+mm]*W_skip[((l*32+k)*Z + z)*32 + c];
    AL[c*16+mm] = acc;
  }
  __syncthreads();
  float nf0=0.f, nf1=0.f, nf2=0.f, nf3=0.f;
  for (int k = 0; k < 32; ++k){
    nf0 += AL[k*16+0]*ccL[k]*Wp_lin[k*32+c];
    float cb = ccL[32+k];
    float wl = Wp_lin[1024 + k*32 + c];
    nf1 += AL[k*16+1]*cb*wl;
    nf2 += AL[k*16+2]*cb*wl;
    nf3 += AL[k*16+3]*cb*wl;
  }
  size_t nfo = (size_t)n*128 + (size_t)c*4;
  outNF[nfo+0]=nf0; outNF[nfo+1]=nf1; outNF[nfo+2]=nf2; outNF[nfo+3]=nf3;
  redL[c] = nf0 * W_read[c];
  __syncthreads();
  if (c == 0){
    float es = 0.f;
    for (int j = 0; j < 32; ++j) es += redL[j];
    float ni = SCALE_C*es + SHIFT_C;
    float e0 = nodeE0[n];
    outNodeE[n] = e0 + ni;
    int g = batch[n];
    atomicAdd(&outInter[g], ni);
    atomicAdd(&outTotal[g], e0 + ni);
  }
  float t = 0.f;
  for (int j = 0; j < 32; ++j) t += Wp_lin[c*32+j]*W_read[j];
  t1L[c] = ccL[c]*t;
  __syncthreads();
  float t2 = 0.f;
  for (int k = 0; k < 32; ++k) t2 += W_skip[((0*32+c)*Z + z)*32 + k]*t1L[k];
  t2L[c] = t2;
  __syncthreads();
  float g = 0.f;
  for (int kp = 0; kp < 32; ++kp) g += W_lin[c*32+kp]*t2L[kp];
  gmsg0[(size_t)n*32 + c] = (SCALE_C/AVG_NEIGHF)*g;
}

// Per-edge backward (wave per edge): recompute MLP forward, backprop through
// rW3 m=0 columns / rW2^T / rW1^T / rW0^T and bessel derivative -> forces.
__global__ void edge_backward(const float* __restrict__ pos, const float* __restrict__ shifts,
                              const int* __restrict__ ei,
                              const float* __restrict__ rW0, const float* __restrict__ rW1,
                              const float* __restrict__ rW2, const float* __restrict__ w3c,
                              const float* __restrict__ nodeUp, const float* __restrict__ gmsg0,
                              float* __restrict__ forces, int E){
  extern __shared__ char smem[];
  float* s_w0  = (float*)smem;       // 512   @ byte 0
  float* s_w1  = s_w0 + 512;         // 4096  @ byte 2048
  float* s_w2  = s_w1 + 4096;        // 4096  @ byte 18432
  float* s_w3c = s_w2 + 4096;        // 2048  @ byte 34816
  float* wsv   = s_w3c + 2048;
  const int wave = threadIdx.x >> 5, lane = threadIdx.x & 31;
  float* efL  = wsv + wave*640;
  float* defL = efL + 8;
  float* h0p  = defL + 8;  float* h0s = h0p + 64;
  float* h1p  = h0s + 64;  float* h1s = h1p + 64;
  float* h2p  = h1s + 64;
  float* g2   = h2p + 64;  float* g1  = g2 + 64;  float* g0 = g1 + 64;
  float* gtpL = g0 + 64;   float* redL = gtpL + 32;

  // async-stage all backward weights (pure byte copies)
  for (int i = threadIdx.x; i < 128;  i += blockDim.x) async_copy_b128(rW0 + i*4, 0u     + (unsigned)i*16);
  for (int i = threadIdx.x; i < 1024; i += blockDim.x) async_copy_b128(rW1 + i*4, 2048u  + (unsigned)i*16);
  for (int i = threadIdx.x; i < 1024; i += blockDim.x) async_copy_b128(rW2 + i*4, 18432u + (unsigned)i*16);
  for (int i = threadIdx.x; i < 512;  i += blockDim.x) async_copy_b128(w3c + i*4, 34816u + (unsigned)i*16);
  async_wait();
  __syncthreads();

  int e = blockIdx.x*BW_WAVES + wave;
  bool ev = (e < E); int ec = ev ? e : (E-1);
  int snd = ei[ec], rcv = ei[E + ec];
  float vx = pos[(long)rcv*3+0]-pos[(long)snd*3+0]+shifts[(long)ec*3+0];
  float vy = pos[(long)rcv*3+1]-pos[(long)snd*3+1]+shifts[(long)ec*3+1];
  float vz = pos[(long)rcv*3+2]-pos[(long)snd*3+2]+shifts[(long)ec*3+2];
  float r  = fmaxf(sqrtf(vx*vx+vy*vy+vz*vz), 1e-6f);
  float inv = 1.f/r;
  float x = r*(1.f/R_MAXF);

  if (lane < 8){
    float cut = 0.f, dcut = 0.f;
    if (x < 1.f){
      float x4 = x*x*x*x, x5 = x4*x, x6 = x5*x;
      cut  = 1.f - 21.f*x5 + 35.f*x6 - 15.f*x6*x;
      dcut = (-105.f*x4 + 210.f*x5 - 105.f*x6)*(1.f/R_MAXF);
    }
    const float C = 0.632455532f;
    float aj = (float)(lane+1)*PI_F*(1.f/R_MAXF);
    float s = __sinf(aj*r), co = __cosf(aj*r);
    float rb = C*s*inv;
    efL[lane]  = rb*cut;
    defL[lane] = C*(aj*co*inv - s*inv*inv)*cut + rb*dcut;
  }
  __syncthreads();
  for (int t = lane; t < 64; t += 32){
    float acc = 0.f;
    for (int k = 0; k < 8; ++k) acc += efL[k]*s_w0[k*64+t];
    h0p[t] = acc; h0s[t] = siluf(acc);
  }
  __syncthreads();
  for (int t = lane; t < 64; t += 32){
    float acc = 0.f;
    for (int k = 0; k < 64; ++k) acc += h0s[k]*s_w1[k*64+t];
    h1p[t] = acc; h1s[t] = siluf(acc);
  }
  __syncthreads();
  for (int t = lane; t < 64; t += 32){
    float acc = 0.f;
    for (int k = 0; k < 64; ++k) acc += h1s[k]*s_w2[k*64+t];
    h2p[t] = acc;
  }
  gtpL[lane] = gmsg0[(size_t)rcv*32 + lane]*nodeUp[(size_t)snd*32 + lane];
  __syncthreads();
  for (int t = lane; t < 64; t += 32){
    float acc = 0.f;
    for (int k = 0; k < 32; ++k) acc += gtpL[k]*s_w3c[t*32+k];
    g2[t] = acc*dsiluf(h2p[t]);
  }
  __syncthreads();
  for (int t = lane; t < 64; t += 32){
    float acc = 0.f;
    for (int k = 0; k < 64; ++k) acc += s_w2[t*64+k]*g2[k];
    g1[t] = acc*dsiluf(h1p[t]);
  }
  __syncthreads();
  for (int t = lane; t < 64; t += 32){
    float acc = 0.f;
    for (int k = 0; k < 64; ++k) acc += s_w1[t*64+k]*g1[k];
    g0[t] = acc*dsiluf(h0p[t]);
  }
  __syncthreads();
  if (lane < 8){
    float acc = 0.f;
    for (int k = 0; k < 64; ++k) acc += s_w0[lane*64+k]*g0[k];
    redL[lane] = acc*defL[lane];
  }
  __syncthreads();
  if (lane == 0){
    float d = 0.f;
    for (int j = 0; j < 8; ++j) d += redL[j];
    redL[16] = d;
  }
  __syncthreads();
  if (ev && lane < 3){
    float dEdr = redL[16];
    float v3 = (lane==0) ? vx : ((lane==1) ? vy : vz);
    float dv = dEdr * v3 * inv;   // dE/dpos component
    atomicAdd(&forces[(size_t)rcv*3 + lane], -dv);  // forces = -grad
    atomicAdd(&forces[(size_t)snd*3 + lane],  dv);
  }
}

// ------------------------------- launch ------------------------------------

extern "C" void kernel_launch(void* const* d_in, const int* in_sizes, int n_in,
                              void* d_out, int out_size, void* d_ws, size_t ws_size,
                              hipStream_t stream) {
  (void)n_in; (void)ws_size;
  const float* positions       = (const float*)d_in[0];
  const float* node_attrs      = (const float*)d_in[1];
  const float* shifts          = (const float*)d_in[2];
  const float* atomic_energies = (const float*)d_in[3];
  const float* W_embed         = (const float*)d_in[4];
  const float* rW0             = (const float*)d_in[5];
  const float* rW1             = (const float*)d_in[6];
  const float* rW2             = (const float*)d_in[7];
  const float* rW3             = (const float*)d_in[8];
  const float* W_up            = (const float*)d_in[9];
  const float* W_lin           = (const float*)d_in[10];
  const float* W_skip          = (const float*)d_in[11];
  const float* Wp_c            = (const float*)d_in[12];
  const float* Wp_lin          = (const float*)d_in[13];
  const float* W_read          = (const float*)d_in[14];
  const int*   edge_index      = (const int*)d_in[15];
  const int*   batch           = (const int*)d_in[16];

  const int N = in_sizes[0] / 3;
  const int E = in_sizes[15] / 2;
  const int Z = in_sizes[1] / N;
  const int G = (out_size - 132*N) / 2;   // out = G + N + G + 3N + 128N

  float* out = (float*)d_out;
  float* outTotal  = out;
  float* outNodeE  = out + G;
  float* outInter  = out + G + N;
  float* outForces = out + 2*G + N;
  float* outNF     = out + 2*G + 4*N;

  float* wsf    = (float*)d_ws;
  float* nodeUp = wsf;
  float* gmsg0  = nodeUp + (size_t)N*32;
  float* nodeE0 = gmsg0  + (size_t)N*32;
  float* msg    = nodeE0 + N;
  int*   zidx   = (int*)(msg + (size_t)N*512);
  size_t wbase  = (size_t)N*32*2 + N + (size_t)N*512 + N;
  wbase = (wbase + 7) & ~(size_t)7;       // 32B align
  float*    w3c = wsf + wbase;            // 2048 f32
  _Float16* wH  = (_Float16*)(w3c + 2048);// 18432 f16 (32B aligned)

  zero_f32<<<1024, 256, 0, stream>>>(msg, (long)N*512);
  zero_f32<<<64,  256, 0, stream>>>(outForces, (long)3*N);
  zero_f32<<<1,    64, 0, stream>>>(outTotal, G);
  zero_f32<<<1,    64, 0, stream>>>(outInter, G);

  prep_weights<<<64, 256, 0, stream>>>(rW0, rW1, rW2, rW3, wH, w3c);

  node_prep<<<N, 32, 0, stream>>>(node_attrs, W_embed, W_up, atomic_energies,
                                  nodeUp, zidx, nodeE0, N, Z);

  int tiles   = (E + TILE - 1) / TILE;
  int fblocks = (tiles + WPB - 1) / WPB;
  size_t ldsF = (size_t)WPB*2816*4 + (size_t)WPB*16*4
              + (size_t)(WH_TOTAL + WPB*2*16*64)*2;
  edge_forward<<<fblocks, 32*WPB, ldsF, stream>>>(positions, shifts, edge_index,
                                                  wH, nodeUp, msg, E);

  node_update<<<N, 32, 0, stream>>>(msg, zidx, nodeE0, W_lin, W_skip, Wp_c, Wp_lin,
                                    W_read, batch, outTotal, outNodeE, outInter,
                                    outNF, gmsg0, N, Z);

  int bblocks = (E + BW_WAVES - 1) / BW_WAVES;
  size_t ldsB = (size_t)(512 + 4096 + 4096 + 2048 + BW_WAVES*640) * 4;
  edge_backward<<<bblocks, 32*BW_WAVES, ldsB, stream>>>(positions, shifts, edge_index,
                                                        rW0, rW1, rW2, w3c, nodeUp,
                                                        gmsg0, outForces, E);
}
